// M3Loss_91130616086868
// MI455X (gfx1250) — matrix operand order repaired and checked by reading, not compiled, not run on hardware
//
#include <hip/hip_runtime.h>
#include <hip/hip_bf16.h>
#include <math.h>

// ---------------------------------------------------------------------------
// M3 fusion loss on gfx1250 (MI455X).
// Separable 11x11 Gaussian / 9x9 box convolutions run as
// V_WMMA_F32_16X16X32_F16 matmuls against banded Toeplitz filter fragments.
// fp32 halo tiles live in LDS; the row-pass intermediate is stored f16
// *transposed* so both the WMMA D write-back (1x ds_store_b128/lane) and the
// col-pass B-fragment gather (2x ds_load_b128/lane) are fully vectorized and
// bank-conflict-free (stride 56 f16 = 28 banks -> gap-4 tiling of 64 banks).
// Global staging uses 16B-aligned float4 loads (left halo = 8; the banded
// filter matrix absorbs the 3-column shift).
// ---------------------------------------------------------------------------

typedef __attribute__((ext_vector_type(16))) _Float16 v16h;
typedef __attribute__((ext_vector_type(8)))  _Float16 v8h;
typedef __attribute__((ext_vector_type(8)))  float    v8f;

#define IMG_H   768
#define IMG_W   768
#define BATCH   16
#define TILE_Y  32          // output rows per workgroup
#define TILE_X  64          // output cols per workgroup
#define SROWS   48          // staged rows (TILE_Y + 2*5 halo, rounded to 3x16)
#define SSTRIDE 84          // staged cols (left halo 8, right slack; 21 float4)
#define TTS     56          // transposed intermediate: row-dim stride (48 + pad)
#define NTOTF   (16.0f * 768.0f * 768.0f)

// band weight; gauss taps t in [0,10], box taps t in [1,9], zero outside.
__device__ __forceinline__ float bandW(int t, bool box, float gnorm) {
  if (box) { int u = t - 1; return (u >= 0 && u <= 8) ? (1.0f / 9.0f) : 0.0f; }
  if (t < 0 || t > 10) return 0.0f;
  float d = (float)(t - 5);
  return expf(-d * d * (1.0f / 4.5f)) * gnorm;   // sigma = 1.5
}

__global__ void m3_zero(unsigned* ws) {
  int i = blockIdx.x * blockDim.x + threadIdx.x;
  if (i < 2304) ws[i] = 0u;    // 84 float accumulators + pad + 2048-word bitmap
}

__global__ __launch_bounds__(256)
void m3_main(const float* __restrict__ outI, const float* __restrict__ viI,
             const float* __restrict__ irI, float* __restrict__ F,
             unsigned* __restrict__ BM) {
  __shared__ __align__(16) float    sIn[3 * SROWS * SSTRIDE];  // 48384 B
  __shared__ __align__(16) _Float16 sTmpT[TILE_X * TTS];       //  7168 B
  __shared__ unsigned               sBM[2048];                 //  8192 B

  const int tid  = threadIdx.x;
  const int wave = tid >> 5;
  const int lane = tid & 31;
  const int hh   = lane >> 4;      // lane half (0/1)
  const int ln   = lane & 15;
  const int bz   = blockIdx.z;
  const int y0   = blockIdx.y * TILE_Y;
  const int x0   = blockIdx.x * TILE_X;
  const int tyI  = wave >> 2;      // wave's 16x16 tile row (0..1)
  const int txI  = wave & 3;       // wave's 16x16 tile col (0..3)

  for (int i = tid; i < 2048; i += 256) sBM[i] = 0u;

  // ---- stage 3 halo tiles: aligned float4 groups, SAME zero padding ----
  // sIn col c <-> global x = x0 - 8 + c   (x0 mult of 64 => 16B aligned)
  // sIn row r <-> global y = y0 - 5 + r
  const int NG = SSTRIDE / 4;   // 21 groups per row
  for (int i = tid; i < 3 * SROWS * NG; i += 256) {
    int s   = i / (SROWS * NG);
    int rem = i - s * (SROWS * NG);
    int r   = rem / NG;
    int g   = rem - r * NG;
    int y   = y0 - 5 + r;
    int x   = x0 - 8 + 4 * g;     // 4-aligned: group fully in or fully out
    float4 val = make_float4(0.f, 0.f, 0.f, 0.f);
    if (y >= 0 && y < IMG_H && x >= 0 && x < IMG_W) {
      const float* sp = (s == 0) ? outI : (s == 1) ? viI : irI;
      val = *(const float4*)&sp[(size_t)(bz * IMG_H + y) * IMG_W + x];
    }
    *(float4*)&sIn[s * (SROWS * SSTRIDE) + r * SSTRIDE + 4 * g] = val;
  }

  // ---- Gaussian normalization ----
  float gs = 0.0f;
#pragma unroll
  for (int u = 0; u < 11; ++u) { float du = (float)(u - 5); gs += expf(-du * du * (1.0f / 4.5f)); }
  const float gnorm = 1.0f / gs;

  // ---- constant banded filter fragments ----
  // Row pass, filter in B (32x16): lane l, vgpr v holds K = 2v + 16*(l>>4)
  //   and K+1; column N = l&15.  B[k][n] = band(k - n - 3)  (x halo = 8).
  // Col pass, filter in A (16x32): lane l = row M, vgpr v holds
  //   K = 2v + 16*(l>>4) and K+1.  A[m][k] = band(k - m) (y halo = 5).
  v16h bfR[2], afC[2];   // [0]=gauss11, [1]=box9
#pragma unroll
  for (int ft = 0; ft < 2; ++ft) {
#pragma unroll
    for (int v = 0; v < 8; ++v) {
      int kb = 2 * v + 16 * hh;
      bfR[ft][2 * v]     = (_Float16)bandW(kb     - ln - 3, ft != 0, gnorm);
      bfR[ft][2 * v + 1] = (_Float16)bandW(kb + 1 - ln - 3, ft != 0, gnorm);
      int ka = 2 * v + 16 * hh;
      afC[ft][2 * v]     = (_Float16)bandW(ka     - ln, ft != 0, gnorm);
      afC[ft][2 * v + 1] = (_Float16)bandW(ka + 1 - ln, ft != 0, gnorm);
    }
  }

  // 11 filtered maps: gauss(o,v,i,o2,v2,i2,ov,oi), box(o,v,i)
  const int srcA[11] = {0, 1, 2, 0, 1, 2, 0, 0, 0, 1, 2};
  const int srcB[11] = {-1, -1, -1, 0, 1, 2, 1, 2, -1, -1, -1};
  const int isB[11]  = {0, 0, 0, 0, 0, 0, 0, 0, 1, 1, 1};

  auto fetch = [&](int map, int r, int c) -> float {
    float a = sIn[srcA[map] * (SROWS * SSTRIDE) + r * SSTRIDE + c];
    int sb = srcB[map];
    if (sb >= 0) a *= sIn[sb * (SROWS * SSTRIDE) + r * SSTRIDE + c];
    return a;
  };

  v8f res[11];
  const v8f cz = {};

#pragma unroll
  for (int map = 0; map < 11; ++map) {
    __syncthreads();  // sTmpT free (prev col pass done) & sIn staged
    // ---- row pass: 12 (3 row x 4 col) 16x16 tiles over 48x64 region ----
    for (int tile = wave; tile < 12; tile += 8) {
      int rt = tile >> 2, ct = tile & 3;
      v16h a;
      int row = 16 * rt + ln;
#pragma unroll
      for (int v = 0; v < 8; ++v) {
        int k = 2 * v + 16 * hh;   // even => b128
        a[2 * v]     = (_Float16)fetch(map, row, 16 * ct + k);
        a[2 * v + 1] = (_Float16)fetch(map, row, 16 * ct + k + 1);
      }
      v8f d = __builtin_amdgcn_wmma_f32_16x16x32_f16(
          false, a, false, bfR[isB[map]], (short)0, cz, false, false);
      // D element j: tmp(row 16rt + j + 8hh, col 16ct + ln) -> transposed,
      // 8 consecutive f16 per lane, 16B aligned -> one ds_store_b128.
      v8h pk;
#pragma unroll
      for (int j = 0; j < 8; ++j) pk[j] = (_Float16)d[j];
      *(v8h*)&sTmpT[(16 * ct + ln) * TTS + 16 * rt + 8 * hh] = pk;
    }
    __syncthreads();
    // ---- col pass: data in B; 16 consecutive f16 per lane -> 2x b128 ----
    int cb = (16 * txI + ln) * TTS + 16 * tyI + 16 * hh;
    v8h b0 = *(const v8h*)&sTmpT[cb];
    v8h b1 = *(const v8h*)&sTmpT[cb + 8];
    v16h bfrag;
#pragma unroll
    for (int e = 0; e < 8; ++e) { bfrag[e] = b0[e]; bfrag[8 + e] = b1[e]; }
    res[map] = __builtin_amdgcn_wmma_f32_16x16x32_f16(
        false, afC[isB[map]], false, bfrag, (short)0, cz, false, false);
  }

  // ---- pointwise phase (fp32 from LDS; res[] layout M=j+8*hh, N=ln) ----
  float accM = 0.f, accG = 0.f, accSV = 0.f, accSI = 0.f;
  float aOV = 0.f, aOI = 0.f, aOO = 0.f, aVV = 0.f, aII = 0.f;
  const float C1 = 1e-4f, C2 = 9e-4f;
#pragma unroll
  for (int j = 0; j < 8; ++j) {
    int m = j + 8 * hh;
    int r = 5 + 16 * tyI + m;
    int c = 8 + 16 * txI + ln;
    float p = sIn[0 * (SROWS * SSTRIDE) + r * SSTRIDE + c];
    float v = sIn[1 * (SROWS * SSTRIDE) + r * SSTRIDE + c];
    float q = sIn[2 * (SROWS * SSTRIDE) + r * SSTRIDE + c];
    float tgt = fmaxf(v, q);
    float dm = p - tgt;
    accM += dm * dm;
    // Sobel gradient magnitude + gamma
    float gm[3];
#pragma unroll
    for (int s = 0; s < 3; ++s) {
      const float* S = &sIn[s * (SROWS * SSTRIDE)];
      float a00 = S[(r - 1) * SSTRIDE + c - 1], a01 = S[(r - 1) * SSTRIDE + c],
            a02 = S[(r - 1) * SSTRIDE + c + 1];
      float a10 = S[r * SSTRIDE + c - 1], a12 = S[r * SSTRIDE + c + 1];
      float a20 = S[(r + 1) * SSTRIDE + c - 1], a21 = S[(r + 1) * SSTRIDE + c],
            a22 = S[(r + 1) * SSTRIDE + c + 1];
      float gx = (a02 + 2.f * a12 + a22) - (a00 + 2.f * a10 + a20);
      float gy = (a00 + 2.f * a01 + a02) - (a20 + 2.f * a21 + a22);
      float g = sqrtf(gx * gx + gy * gy + 1e-10f);
      gm[s] = powf(g + 1e-10f, 0.8f);
    }
    accG += fabsf(gm[0] - fmaxf(gm[1], gm[2]));
    // SSIM maps
    float mo = res[0][j], mv = res[1][j], mq = res[2][j];
    float moo = res[3][j], mvv = res[4][j], mqq = res[5][j];
    float mov = res[6][j], moq = res[7][j];
    {
      float s1 = moo - mo * mo, s2 = mvv - mv * mv, s12 = mov - mo * mv;
      accSV += ((2.f * mo * mv + C1) * (2.f * s12 + C2)) /
               ((mo * mo + mv * mv + C1) * (s1 + s2 + C2));
    }
    {
      float s1 = moo - mo * mo, s2 = mqq - mq * mq, s12 = moq - mo * mq;
      accSI += ((2.f * mo * mq + C1) * (2.f * s12 + C2)) /
               ((mo * mo + mq * mq + C1) * (s1 + s2 + C2));
    }
    // NCC z-products (box means in res[8..10])
    float z1 = p - res[8][j], zv = v - res[9][j], zq = q - res[10][j];
    aOV += z1 * zv; aOI += z1 * zq; aOO += z1 * z1; aVV += zv * zv; aII += zq * zq;
    // MI occupancy bits
    int bx = (int)(p * 255.0f);   bx = bx < 0 ? 0 : (bx > 255 ? 255 : bx);
    int by = (int)(tgt * 255.0f); by = by < 0 ? 0 : (by > 255 ? 255 : by);
    int bin = (bx << 8) | by;
    atomicOr(&sBM[bin >> 5], 1u << (bin & 31));
  }

  // ---- wave reduce + global accumulation ----
  auto wred = [](float x) {
#pragma unroll
    for (int o = 16; o > 0; o >>= 1) x += __shfl_down(x, o, 32);
    return x;
  };
  accM = wred(accM); accG = wred(accG); accSV = wred(accSV); accSI = wred(accSI);
  aOV = wred(aOV); aOI = wred(aOI); aOO = wred(aOO); aVV = wred(aVV); aII = wred(aII);
  if (lane == 0) {
    atomicAdd(&F[0], accM);  atomicAdd(&F[1], accG);
    atomicAdd(&F[2], accSV); atomicAdd(&F[3], accSI);
    atomicAdd(&F[4 + bz],  aOV); atomicAdd(&F[20 + bz], aOI);
    atomicAdd(&F[36 + bz], aOO); atomicAdd(&F[52 + bz], aVV);
    atomicAdd(&F[68 + bz], aII);
  }
  __syncthreads();
  for (int w = tid; w < 2048; w += 256) {
    unsigned b = sBM[w];
    if (b) atomicOr(&BM[w], b);
  }
}

__global__ __launch_bounds__(256)
void m3_finalize(const float* __restrict__ F, const unsigned* __restrict__ BM,
                 float* __restrict__ out) {
  __shared__ unsigned lbm[2048];
  __shared__ float rowc[256], colc[256], red[256];
  __shared__ float Tsh;
  int tid = threadIdx.x;
#pragma unroll
  for (int w = 0; w < 8; ++w) lbm[tid * 8 + w] = BM[tid * 8 + w];
  __syncthreads();
  int rc = 0;
#pragma unroll
  for (int w = 0; w < 8; ++w) rc += __popc(lbm[tid * 8 + w]);
  rowc[tid] = (float)rc;
  int cc = 0, wsel = tid >> 5, bsel = tid & 31;
  for (int i = 0; i < 256; ++i) cc += (int)((lbm[i * 8 + wsel] >> bsel) & 1u);
  colc[tid] = (float)cc;
  __syncthreads();
  if (tid == 0) { float t = 0.f; for (int i = 0; i < 256; ++i) t += rowc[i]; Tsh = t; }
  __syncthreads();
  float T = Tsh;            // >= 1 always
  float h = 1.0f / T;
  float px = rowc[tid] * h;
  float term = 0.f;
  for (int w = 0; w < 8; ++w) {
    unsigned word = lbm[tid * 8 + w];
    while (word) {
      int b = __ffs(word) - 1;
      word &= word - 1;
      float py = colc[w * 32 + b] * h;
      term += h * logf(h / (px * py + 1e-10f));
    }
  }
  red[tid] = term;
  __syncthreads();
  for (int off = 128; off > 0; off >>= 1) {
    if (tid < off) red[tid] += red[tid + off];
    __syncthreads();
  }
  if (tid == 0) {
    float miL = 1.0f - tanhf(red[0]);
    float nccOV = 0.f, nccOI = 0.f;
    for (int b = 0; b < 16; ++b) {
      float sov = F[4 + b], soi = F[20 + b];
      float soo = F[36 + b], svv = F[52 + b], sii = F[68 + b];
      nccOV += 1.0f - sov / (sqrtf(soo * svv) + 1e-8f);
      nccOI += 1.0f - soi / (sqrtf(soo * sii) + 1e-8f);
    }
    nccOV *= (1.0f / 16.0f);
    nccOI *= (1.0f / 16.0f);
    float inv = 1.0f / NTOTF;
    float mse = F[0] * inv, grad = F[1] * inv;
    float ssimOV = 1.0f - F[2] * inv, ssimOI = 1.0f - F[3] * inv;
    out[0] = 10.0f * mse + 10.0f * grad + (ssimOV + ssimOI) +
             (nccOV + nccOI) + miL;
  }
}

extern "C" void kernel_launch(void* const* d_in, const int* in_sizes, int n_in,
                              void* d_out, int out_size, void* d_ws, size_t ws_size,
                              hipStream_t stream) {
  (void)in_sizes; (void)n_in; (void)out_size; (void)ws_size;
  const float* o = (const float*)d_in[0];
  const float* v = (const float*)d_in[1];
  const float* r = (const float*)d_in[2];
  float*    F  = (float*)d_ws;                         // 84 accumulators
  unsigned* BM = (unsigned*)((char*)d_ws + 1024);      // 2048-word bitmap
  m3_zero<<<9, 256, 0, stream>>>((unsigned*)d_ws);
  dim3 grid(IMG_W / TILE_X, IMG_H / TILE_Y, BATCH);    // 12 x 24 x 16
  m3_main<<<grid, 256, 0, stream>>>(o, v, r, F, BM);
  m3_finalize<<<1, 256, 0, stream>>>(F, BM, (float*)d_out);
}
